// SplitLayer0_1_10462540333326
// MI455X (gfx1250) — compile-verified, matrix operand order checked by reading, not run on hardware
//
#include <hip/hip_runtime.h>

typedef __attribute__((ext_vector_type(16))) __bf16 v16bf;
typedef __attribute__((ext_vector_type(8)))  float  v8f;

__device__ __forceinline__ unsigned short f32_to_bf16(float f) {
  unsigned int u = __float_as_uint(f);
  u += 0x7fffu + ((u >> 16) & 1u);          // round to nearest even
  return (unsigned short)(u >> 16);
}

// ---------------------------------------------------------------------------
// Generic bf16 WMMA GEMM:  C[M x Nout] = A[M x K](bf16) * W[K x Nout](bf16)
// Block: 256 threads = 8 waves. Block tile 128(M) x 64(N); wave tile 16x64:
// one A-fragment is reused across 4 WMMAs (4x less A traffic from HBM).
// Weight panel (K x 64) staged transposed in LDS -> conflict-free ds_load_b128.
// ---------------------------------------------------------------------------
__global__ __launch_bounds__(256)
void gemm_bf16_wmma_kernel(const unsigned short* __restrict__ A,
                           const unsigned short* __restrict__ W,
                           float* __restrict__ C,
                           int M, int K, int Nout) {
  constexpr int BS_STRIDE = 392;            // max K (384) + 8 pad (bank spread)
  __shared__ __align__(16) unsigned short Bs[64 * BS_STRIDE];   // ~49 KB

  const int n0 = blockIdx.y * 64;
  for (int idx = threadIdx.x; idx < 64 * K; idx += 256) {
    int n = idx / K;
    int k = idx - n * K;
    Bs[n * BS_STRIDE + k] = W[(size_t)k * Nout + n0 + n];
  }
  __syncthreads();

  const int lane = threadIdx.x & 31;
  const int wid  = threadIdx.x >> 5;
  const int m0   = blockIdx.x * 128 + wid * 16;

  int arow = m0 + (lane & 15);
  if (arow >= M) arow = M - 1;              // clamp; OOB rows masked on store
  const int khalf = (lane & 16) ? 8 : 0;    // wave32 16-bit A/B VGPR layout
  const unsigned short* Aptr = A + (size_t)arow * K + khalf;
  const unsigned short* Bp   = Bs + (size_t)(lane & 15) * BS_STRIDE + khalf;

  const v8f vzero = {0.f, 0.f, 0.f, 0.f, 0.f, 0.f, 0.f, 0.f};
  v8f acc[4] = {vzero, vzero, vzero, vzero};

  for (int kk = 0; kk < K; kk += 32) {
    union { uint4 u[2]; v16bf v; } afrag;
    afrag.u[0] = *(const uint4*)(Aptr + kk);        // K kk..kk+7   (this half)
    afrag.u[1] = *(const uint4*)(Aptr + kk + 16);   // K kk+16..23  (this half)
    if (kk + 32 < K)
      __builtin_prefetch((const void*)(Aptr + kk + 32), 0, 1);
#pragma unroll
    for (int j = 0; j < 4; ++j) {
      union { uint4 u[2]; v16bf v; } bfrag;
      const unsigned short* bp = Bp + (size_t)j * 16 * BS_STRIDE + kk;
      bfrag.u[0] = *(const uint4*)(bp);
      bfrag.u[1] = *(const uint4*)(bp + 16);
      acc[j] = __builtin_amdgcn_wmma_f32_16x16x32_bf16(
          false, afrag.v, false, bfrag.v, (short)0, acc[j], false, false);
    }
  }

  const int mbase = m0 + ((lane & 16) ? 8 : 0);     // f32 C/D layout
#pragma unroll
  for (int j = 0; j < 4; ++j) {
    const int n = n0 + j * 16 + (lane & 15);
#pragma unroll
    for (int i = 0; i < 8; ++i) {
      int m = mbase + i;
      if (m < M) C[(size_t)m * Nout + n] = acc[j][i];
    }
  }
}

// ---------------------------------------------------------------------------
// Builders / reductions / elementwise
// ---------------------------------------------------------------------------
__global__ void f32_to_bf16_kernel(const float* __restrict__ src,
                                   unsigned short* __restrict__ dst, int n) {
  int i = blockIdx.x * blockDim.x + threadIdx.x;
  if (i < n) dst[i] = f32_to_bf16(src[i]);
}

// X1[r] = [ node[u]+node[v] | node[atom(r)] | edge_rep[r] ]   (bf16, 384 cols)
__global__ void build_x1_kernel(const float* __restrict__ node_rep,
                                const float* __restrict__ edge_rep,
                                const long long* __restrict__ ei,
                                unsigned short* __restrict__ X1, int Ecnt) {
  const int r = blockIdx.x, c = threadIdx.x;
  const int e = r >> 1;
  const int u = (int)ei[e];
  const int v = (int)ei[Ecnt + e];
  const float du = node_rep[(size_t)u * 128 + c];
  const float dv = node_rep[(size_t)v * 128 + c];
  unsigned short* row = X1 + (size_t)r * 384;
  row[c]       = f32_to_bf16(du + dv);
  row[128 + c] = f32_to_bf16((r & 1) ? dv : du);
  row[256 + c] = f32_to_bf16(edge_rep[(size_t)r * 128 + c]);
}

// lvl_local[a] += h[r]; lvl_dom[a] += h[2e]+h[2e+1]
__global__ void scatter_kernel(const float* __restrict__ h,
                               const long long* __restrict__ ei,
                               float* __restrict__ lvl_local,
                               float* __restrict__ lvl_dom, int Ecnt) {
  const int r = blockIdx.x, c = threadIdx.x;
  const int e = r >> 1;
  const int u = (int)ei[e];
  const int v = (int)ei[Ecnt + e];
  const int a = (r & 1) ? v : u;
  const float hv = h[(size_t)r * 128 + c];
  const float ho = h[(size_t)(r ^ 1) * 128 + c];
  atomicAdd(&lvl_local[(size_t)a * 128 + c], hv);
  atomicAdd(&lvl_dom[(size_t)a * 128 + c], hv + ho);
}

__global__ void build_node_in_kernel(const float* __restrict__ node_rep,
                                     const float* __restrict__ lvl_local,
                                     const float* __restrict__ lvl_dom,
                                     const float* eps11p, const float* eps12p,
                                     unsigned short* __restrict__ out,
                                     long long total) {
  const float a = 1.f + *eps11p, b = 1.f + *eps12p;
  for (long long i = blockIdx.x * (long long)blockDim.x + threadIdx.x;
       i < total; i += (long long)gridDim.x * blockDim.x)
    out[i] = f32_to_bf16(a * node_rep[i] + b * lvl_local[i] + lvl_dom[i]);
}

// edge_in[r] = (1+eps2)*[edge_mean | edge_rep[r]] + [node[u]+node[v] | node[atom]]
__global__ void build_aedge_kernel(const float* __restrict__ node_rep,
                                   const float* __restrict__ edge_rep,
                                   const long long* __restrict__ ei,
                                   const float* eps2p,
                                   unsigned short* __restrict__ Aed, int Ecnt) {
  const int r = blockIdx.x, c = threadIdx.x;
  const int e = r >> 1;
  const int u = (int)ei[e];
  const int v = (int)ei[Ecnt + e];
  const float s2 = 1.f + *eps2p;
  const float du = node_rep[(size_t)u * 128 + c];
  const float dv = node_rep[(size_t)v * 128 + c];
  const float eu = edge_rep[(size_t)(r & ~1) * 128 + c];
  const float ev = edge_rep[(size_t)(r | 1) * 128 + c];
  unsigned short* row = Aed + (size_t)r * 256;
  row[c]       = f32_to_bf16(s2 * 0.5f * (eu + ev) + du + dv);
  row[128 + c] = f32_to_bf16(s2 * ((r & 1) ? ev : eu) + ((r & 1) ? dv : du));
}

// per-column sum & sum-of-squares -> stats[0..C) and stats[C..2C)
__global__ void col_stats_kernel(const float* __restrict__ X,
                                 float* __restrict__ stats,
                                 int Mrows, int C, int rowsPerBlock) {
  const int t    = threadIdx.x;
  const int c    = t & (C - 1);             // C is 128 or 256 (power of two)
  const int nsub = blockDim.x / C;
  const int sub  = t / C;
  long long r0   = (long long)blockIdx.x * rowsPerBlock + sub;
  long long rend = (long long)(blockIdx.x + 1) * rowsPerBlock;
  if (rend > Mrows) rend = Mrows;
  float s = 0.f, sq = 0.f;
  for (long long r = r0; r < rend; r += nsub) {
    float x = X[r * C + c];
    s += x; sq += x * x;
  }
  atomicAdd(&stats[c], s);
  atomicAdd(&stats[C + c], sq);
}

// y = relu((x-mean)*rsqrt(var+eps)*g + b); optional f32 and/or bf16 outputs
__global__ void bn_relu_kernel(const float* __restrict__ X,
                               const float* __restrict__ stats,
                               const float* __restrict__ g,
                               const float* __restrict__ b,
                               float* __restrict__ outF,
                               unsigned short* __restrict__ outB,
                               long long Mrows, int C) {
  const long long total = Mrows * C;
  const float invM = 1.f / (float)Mrows;
  for (long long i = blockIdx.x * (long long)blockDim.x + threadIdx.x;
       i < total; i += (long long)gridDim.x * blockDim.x) {
    int c = (int)(i % C);
    float mean = stats[c] * invM;
    float var  = stats[C + c] * invM - mean * mean;
    float inv  = rsqrtf(var + 1e-5f);
    float y = (X[i] - mean) * inv * g[c] + b[c];
    y = y > 0.f ? y : 0.f;
    if (outF) outF[i] = y;
    if (outB) outB[i] = f32_to_bf16(y);
  }
}

// ---------------------------------------------------------------------------
extern "C" void kernel_launch(void* const* d_in, const int* in_sizes, int n_in,
                              void* d_out, int out_size, void* d_ws, size_t ws_size,
                              hipStream_t stream) {
  (void)in_sizes; (void)n_in; (void)out_size; (void)ws_size;
  const int Nn = 20000, Ee = 160000;
  const int M2E = 2 * Ee;

  const float*     node_rep = (const float*)d_in[0];
  const float*     edge_rep = (const float*)d_in[1];
  const long long* ei       = (const long long*)d_in[2];
  const float* lift_w1 = (const float*)d_in[3];
  const float* lift_g1 = (const float*)d_in[4];
  const float* lift_b1 = (const float*)d_in[5];
  const float* lift_w2 = (const float*)d_in[6];
  const float* lift_g2 = (const float*)d_in[7];
  const float* lift_b2 = (const float*)d_in[8];
  const float* lvl1_w  = (const float*)d_in[9];
  const float* lvl1_g  = (const float*)d_in[10];
  const float* lvl1_b  = (const float*)d_in[11];
  const float* lvl2_w1 = (const float*)d_in[12];
  const float* lvl2_g1 = (const float*)d_in[13];
  const float* lvl2_b1 = (const float*)d_in[14];
  const float* lvl2_w2 = (const float*)d_in[15];
  const float* lvl2_g2 = (const float*)d_in[16];
  const float* lvl2_b2 = (const float*)d_in[17];
  const float* eps11   = (const float*)d_in[18];
  const float* eps12   = (const float*)d_in[19];
  const float* eps2    = (const float*)d_in[20];

  // ---- workspace arena (~560 MB, lifetime-based region reuse) ----
  char* ws = (char*)d_ws;
  size_t off = 0;
  auto take = [&](size_t bytes) -> char* {
    char* p = ws + off;
    off = (off + bytes + 255) & ~(size_t)255;
    return p;
  };
  float*          stats   = (float*)take(2048);
  unsigned short* w_lvl1  = (unsigned short*)take((size_t)384 * 128 * 2);
  unsigned short* w_lvl2a = (unsigned short*)take((size_t)128 * 256 * 2);
  unsigned short* w_lvl2b = (unsigned short*)take((size_t)256 * 128 * 2);
  unsigned short* w_lift1 = (unsigned short*)take((size_t)256 * 256 * 2);
  unsigned short* w_lift2 = (unsigned short*)take((size_t)256 * 128 * 2);
  float*          lvl_local = (float*)take((size_t)Nn * 128 * 4);
  float*          lvl_dom   = (float*)take((size_t)Nn * 128 * 4);
  unsigned short* node_in   = (unsigned short*)take((size_t)Nn * 128 * 2);
  float*          nodeP1    = (float*)take((size_t)Nn * 256 * 4);
  unsigned short* node_h    = (unsigned short*)take((size_t)Nn * 256 * 2);
  float*          nodeP2    = (float*)take((size_t)Nn * 128 * 4);
  char* regionB = take((size_t)M2E * 256 * 2);   // P1(f32) -> Aedge(bf16) -> edge_h(bf16)
  char* regionA = take((size_t)M2E * 256 * 4);   // X1(bf16) -> edgeP1(f32) -> edgeP2(f32)

  float* outNode = (float*)d_out;
  float* outEdge = (float*)d_out + (size_t)Nn * 128;

  // ---- weights -> bf16 ----
  auto cvt = [&](const float* s, unsigned short* d, int n) {
    f32_to_bf16_kernel<<<(n + 255) / 256, 256, 0, stream>>>(s, d, n);
  };
  cvt(lvl1_w,  w_lvl1,  384 * 128);
  cvt(lvl2_w1, w_lvl2a, 128 * 256);
  cvt(lvl2_w2, w_lvl2b, 256 * 128);
  cvt(lift_w1, w_lift1, 256 * 256);
  cvt(lift_w2, w_lift2, 256 * 128);

  // ---- lvl1: h = BN-ReLU( [lift_aggr | edge_rep] @ lvl1_w ) ----
  unsigned short* X1 = (unsigned short*)regionA;
  build_x1_kernel<<<M2E, 128, 0, stream>>>(node_rep, edge_rep, ei, X1, Ee);
  float* P1 = (float*)regionB;
  gemm_bf16_wmma_kernel<<<dim3((M2E + 127) / 128, 2), 256, 0, stream>>>(
      X1, w_lvl1, P1, M2E, 384, 128);
  hipMemsetAsync(stats, 0, 2048, stream);
  col_stats_kernel<<<(M2E + 511) / 512, 256, 0, stream>>>(P1, stats, M2E, 128, 512);
  bn_relu_kernel<<<4096, 256, 0, stream>>>(P1, stats, lvl1_g, lvl1_b,
                                           P1, nullptr, M2E, 128);

  // ---- transfer1_0 scatter ----
  hipMemsetAsync(lvl_local, 0, (size_t)Nn * 128 * 4, stream);
  hipMemsetAsync(lvl_dom,   0, (size_t)Nn * 128 * 4, stream);
  scatter_kernel<<<M2E, 128, 0, stream>>>(P1, ei, lvl_local, lvl_dom, Ee);

  // ---- node branch ----
  build_node_in_kernel<<<2048, 256, 0, stream>>>(node_rep, lvl_local, lvl_dom,
                                                 eps11, eps12, node_in,
                                                 (long long)Nn * 128);
  gemm_bf16_wmma_kernel<<<dim3((Nn + 127) / 128, 4), 256, 0, stream>>>(
      node_in, w_lvl2a, nodeP1, Nn, 128, 256);
  hipMemsetAsync(stats, 0, 2048, stream);
  col_stats_kernel<<<(Nn + 511) / 512, 256, 0, stream>>>(nodeP1, stats, Nn, 256, 512);
  bn_relu_kernel<<<1024, 256, 0, stream>>>(nodeP1, stats, lvl2_g1, lvl2_b1,
                                           nullptr, node_h, Nn, 256);
  gemm_bf16_wmma_kernel<<<dim3((Nn + 127) / 128, 2), 256, 0, stream>>>(
      node_h, w_lvl2b, nodeP2, Nn, 256, 128);
  hipMemsetAsync(stats, 0, 2048, stream);
  col_stats_kernel<<<(Nn + 511) / 512, 256, 0, stream>>>(nodeP2, stats, Nn, 128, 512);
  bn_relu_kernel<<<1024, 256, 0, stream>>>(nodeP2, stats, lvl2_g2, lvl2_b2,
                                           outNode, nullptr, Nn, 128);

  // ---- edge branch ----
  unsigned short* Aedge = (unsigned short*)regionB;   // P1/h dead now
  build_aedge_kernel<<<M2E, 128, 0, stream>>>(node_rep, edge_rep, ei, eps2, Aedge, Ee);
  float* edgeP1 = (float*)regionA;                    // X1 dead now
  gemm_bf16_wmma_kernel<<<dim3((M2E + 127) / 128, 4), 256, 0, stream>>>(
      Aedge, w_lift1, edgeP1, M2E, 256, 256);
  hipMemsetAsync(stats, 0, 2048, stream);
  col_stats_kernel<<<(M2E + 511) / 512, 256, 0, stream>>>(edgeP1, stats, M2E, 256, 512);
  unsigned short* edge_h = (unsigned short*)regionB;  // Aedge dead now
  bn_relu_kernel<<<4096, 256, 0, stream>>>(edgeP1, stats, lift_g1, lift_b1,
                                           nullptr, edge_h, M2E, 256);
  float* edgeP2 = (float*)regionA;                    // edgeP1 dead now
  gemm_bf16_wmma_kernel<<<dim3((M2E + 127) / 128, 2), 256, 0, stream>>>(
      edge_h, w_lift2, edgeP2, M2E, 256, 128);
  hipMemsetAsync(stats, 0, 2048, stream);
  col_stats_kernel<<<(M2E + 511) / 512, 256, 0, stream>>>(edgeP2, stats, M2E, 128, 512);
  bn_relu_kernel<<<4096, 256, 0, stream>>>(edgeP2, stats, lift_g2, lift_b2,
                                           outEdge, nullptr, M2E, 128);
}
